// MultiHeadAttention_6700148982340
// MI455X (gfx1250) — matrix operand order repaired and checked by reading, not compiled, not run on hardware
//
#include <hip/hip_runtime.h>
#include <hip/hip_bf16.h>
#include <math.h>

typedef __attribute__((ext_vector_type(16))) __bf16 v16bf;
typedef __attribute__((ext_vector_type(8)))  float  v8f;

#define NUM_HEADS 16
#define HEAD_DIM  64
#define DIMS      1024
#define BATCH     8
#define SEQ       1024
#define MTOT      (BATCH * SEQ)   // 8192

union FragU { v16bf v; unsigned int w[8]; };

__device__ __forceinline__ unsigned short f2bf(float f) {
  union { float f; unsigned int u; } c; c.f = f;
  unsigned int u = c.u;
  u += 0x7FFFu + ((u >> 16) & 1u);   // round-to-nearest-even
  return (unsigned short)(u >> 16);
}

// A-matrix (16x32 bf16) K-pair base for VGPR word j, lane half (ISA 7.12.2)
__device__ __forceinline__ int a_kmap(int j, int half) {
  return ((j < 4) ? (2 * j) : (16 + 2 * (j - 4))) + 8 * half;
}

// A fragment: M along lanes (lane&15), K pairs per 32-bit word with split-K map
__device__ __forceinline__ v16bf load_frag_a(const unsigned short* base, int row,
                                             int ldk, int kbase, int lane) {
  FragU f;
  const int half = lane >> 4;
  const unsigned short* p = base + row * ldk + kbase;
#pragma unroll
  for (int j = 0; j < 8; ++j)
    f.w[j] = *(const unsigned int*)(p + a_kmap(j, half));
  return f.v;
}

// B fragment: N along lanes (lane&15), K pairs per 32-bit word with split-K map
__device__ __forceinline__ v16bf load_frag_b(const unsigned short* base, int kbase,
                                             int ldn, int col0, int lane) {
  FragU f;
  const int half = lane >> 4;
  const unsigned short* p = base + kbase * ldn + col0 + (lane & 15);
#pragma unroll
  for (int j = 0; j < 8; ++j) {
    const int k = a_kmap(j, half);
    unsigned int lo = p[k * ldn];
    unsigned int hi = p[(k + 1) * ldn];
    f.w[j] = lo | (hi << 16);
  }
  return f.v;
}

__device__ __forceinline__ v8f wmma_bf16(v16bf a, v16bf b, v8f c) {
  return __builtin_amdgcn_wmma_f32_16x16x32_bf16(false, a, false, b,
                                                 (short)0, c, false, false);
}

// CDNA5 async memory->LDS copy (ASYNCcnt-tracked, no VGPR round trip).
// LDS byte offset = low 32 bits of the generic pointer (LDS aperture rule).
__device__ __forceinline__ void async_copy_b128(void* ldsp, const void* gaddr) {
  unsigned off = (unsigned)(size_t)ldsp;
  asm volatile("global_load_async_to_lds_b128 %0, %1, off"
               :: "v"(off), "v"(gaddr) : "memory");
}
__device__ __forceinline__ void wait_async0() {
  asm volatile("s_wait_asynccnt 0x0" ::: "memory");
}

// ---------------------------------------------------------------------------
// Kernel 1: Q/K/V projections. Grid (N/128, M/128, 3). 256 threads = 8 waves,
// wave grid 4(M) x 2(N); each wave a 32x64 tile (8 WMMA accumulators).
// f32 -> bf16 conversion during LDS staging; epilogue scatters to head-major
// layouts (K pre-transposed for the score GEMM's B-fragments).
// ---------------------------------------------------------------------------
__global__ void __launch_bounds__(256) qkv_proj_kernel(
    const float* __restrict__ q, const float* __restrict__ x,
    const float* __restrict__ Wq, const float* __restrict__ bq,
    const float* __restrict__ Wk, const float* __restrict__ bk,
    const float* __restrict__ Wv, const float* __restrict__ bv,
    unsigned short* __restrict__ Qs,  // [b*h][p][d]
    unsigned short* __restrict__ Kt,  // [b*h][d][p]
    unsigned short* __restrict__ Vs)  // [b*h][p][d]
{
  __shared__ __align__(16) unsigned short sA[128][34];
  __shared__ __align__(16) unsigned short sB[32][130];

  const int tid = threadIdx.x, lane = tid & 31, wave = tid >> 5;
  const int wm = wave & 3, wn = wave >> 2;
  const int m0 = blockIdx.y * 128, n0 = blockIdx.x * 128;
  const int which = blockIdx.z;

  const float* A    = (which == 0) ? q  : x;
  const float* W    = (which == 0) ? Wq : (which == 1) ? Wk : Wv;
  const float* bias = (which == 0) ? bq : (which == 1) ? bk : bv;

  v8f acc[2][4] = {};

  for (int k0 = 0; k0 < DIMS; k0 += 32) {
    if (k0 + 32 < DIMS) {   // prefetch next k-tile (global_prefetch)
      __builtin_prefetch(A + (size_t)(m0 + (tid & 127)) * DIMS + k0 + 32, 0, 3);
      __builtin_prefetch(W + (size_t)(k0 + 32 + (tid & 31)) * DIMS + n0, 0, 3);
    }
#pragma unroll
    for (int i = 0; i < 16; ++i) {             // A tile 128x32, f32 -> bf16
      int e = tid + 256 * i;
      int r = e >> 5, c = e & 31;
      sA[r][c] = f2bf(A[(size_t)(m0 + r) * DIMS + k0 + c]);
    }
#pragma unroll
    for (int i = 0; i < 16; ++i) {             // W tile 32x128 row-major [k][n]
      int e = tid + 256 * i;
      int kk = e >> 7, nn = e & 127;
      sB[kk][nn] = f2bf(W[(size_t)(k0 + kk) * DIMS + n0 + nn]);
    }
    __syncthreads();

    v16bf afr[2], bfr[4];
#pragma unroll
    for (int ti = 0; ti < 2; ++ti)
      afr[ti] = load_frag_a(&sA[0][0], wm * 32 + ti * 16 + (lane & 15), 34, 0, lane);
#pragma unroll
    for (int tj = 0; tj < 4; ++tj)
      bfr[tj] = load_frag_b(&sB[0][0], 0, 130, wn * 64 + tj * 16, lane);
#pragma unroll
    for (int ti = 0; ti < 2; ++ti)
#pragma unroll
      for (int tj = 0; tj < 4; ++tj)
        acc[ti][tj] = wmma_bf16(afr[ti], bfr[tj], acc[ti][tj]);
    __syncthreads();
  }

  const int half = lane >> 4, lr = lane & 15;
#pragma unroll
  for (int ti = 0; ti < 2; ++ti) {
#pragma unroll
    for (int tj = 0; tj < 4; ++tj) {
      const int n = n0 + wn * 64 + tj * 16 + lr;
      const float bn = bias[n];
      const int h = n & 15, d = n >> 4;        // reshape (..., hd, H): head is last
#pragma unroll
      for (int r = 0; r < 8; ++r) {
        const int m = m0 + wm * 32 + ti * 16 + r + 8 * half;
        const int bidx = m >> 10, p = m & 1023;
        const int bh = bidx * NUM_HEADS + h;
        const unsigned short v = f2bf(acc[ti][tj][r] + bn);
        if (which == 0)      Qs[((size_t)bh * SEQ + p) * HEAD_DIM + d] = v;
        else if (which == 1) Kt[((size_t)bh * HEAD_DIM + d) * SEQ + p] = v;
        else                 Vs[((size_t)bh * SEQ + p) * HEAD_DIM + d] = v;
      }
    }
  }
}

// ---------------------------------------------------------------------------
// Kernel 2: flash attention, ping-pong pipelined. Block = 64 query rows of
// one (b,h). K/V tiles double-buffered in LDS and staged with
// GLOBAL_LOAD_ASYNC_TO_LDS_B128: tile t+1 streams in (ASYNCcnt) while WMMAs
// and softmax work on tile t. Q fragments are loaded once into registers.
// ---------------------------------------------------------------------------
__global__ void __launch_bounds__(256) attention_kernel(
    const unsigned short* __restrict__ Qs,
    const unsigned short* __restrict__ Kt,
    const unsigned short* __restrict__ Vs,
    unsigned short* __restrict__ A2)   // [b][p][h*64+d]
{
  __shared__ __align__(16) unsigned short sK[2][64][80];  // [buf][d][key]
  __shared__ __align__(16) unsigned short sV[2][64][80];  // [buf][key][d]
  __shared__ float sS[64][65];                            // scores / probs
  __shared__ float sM[64], sL[64], sAl[64];
  __shared__ float sRed[64][4];

  const int tid = threadIdx.x, lane = tid & 31, wave = tid >> 5;
  const int half = lane >> 4, lr = lane & 15;
  const int qi = wave & 3;     // q-row tile owned (rows qi*16..)
  const int dj = wave >> 2;    // output column half (P@V)
  const int kg = wave >> 2;    // key column half (S compute)
  const int srow = tid >> 2, sseg = tid & 3;

  const int qb = blockIdx.x, h = blockIdx.y, b = blockIdx.z;
  const int bh = b * NUM_HEADS + h;
  const int p0 = qb * 64;

  const unsigned short* Qbh = Qs + (size_t)bh * SEQ * HEAD_DIM;
  const unsigned short* Ktb = Kt + (size_t)bh * HEAD_DIM * SEQ;
  const unsigned short* Vbh = Vs + (size_t)bh * SEQ * HEAD_DIM;

  // Q fragments: wave-invariant across the whole loop -> gather once from
  // global straight into the WMMA A-fragment register layout.
  v16bf qfr[2];
  {
    const unsigned short* qp = Qbh + (size_t)(p0 + qi * 16 + lr) * HEAD_DIM;
    FragU f0, f1;
#pragma unroll
    for (int j = 0; j < 8; ++j) {
      f0.w[j] = *(const unsigned int*)(qp + a_kmap(j, half));
      f1.w[j] = *(const unsigned int*)(qp + a_kmap(j, half) + 32);
    }
    qfr[0] = f0.v; qfr[1] = f1.v;
  }
  if (tid < 64) { sM[tid] = -3.0e38f; sL[tid] = 0.f; }

  // Prime the pipeline: stage tile 0 into buffer 0 (512 x 16B per tile pair).
#pragma unroll
  for (int i = 0; i < 2; ++i) {
    int e = tid + 256 * i;
    int r = e >> 3, c8 = (e & 7) * 8;
    async_copy_b128(&sK[0][r][c8], Ktb + (size_t)r * SEQ + c8);
    async_copy_b128(&sV[0][r][c8], Vbh + (size_t)r * HEAD_DIM + c8);
  }

  v8f accO[2] = {};
  int cur = 0;

  for (int kt = 0; kt < SEQ; kt += 64, cur ^= 1) {
    wait_async0();        // buffer `cur` staged (each wave waits its own share)
    __syncthreads();      // visible to all waves; all waves done with `cur^1`

    if (kt + 64 < SEQ) {  // kick off tile t+1 into the other buffer
      const int nb = cur ^ 1;
#pragma unroll
      for (int i = 0; i < 2; ++i) {
        int e = tid + 256 * i;
        int r = e >> 3, c8 = (e & 7) * 8;
        async_copy_b128(&sK[nb][r][c8], Ktb + (size_t)r * SEQ + kt + 64 + c8);
        async_copy_b128(&sV[nb][r][c8], Vbh + (size_t)(kt + 64 + r) * HEAD_DIM + c8);
      }
    }

    // S = (Q @ K^T) / sqrt(64); wave: rows qi*16.., cols kg*32..
#pragma unroll
    for (int tj = 0; tj < 2; ++tj) {
      v8f s = {};
#pragma unroll
      for (int kk = 0; kk < 64; kk += 32) {
        v16bf bb = load_frag_b(&sK[cur][0][0], kk, 80, kg * 32 + tj * 16, lane);
        s = wmma_bf16(qfr[kk >> 5], bb, s);
      }
      const int col = kg * 32 + tj * 16 + lr;
#pragma unroll
      for (int r = 0; r < 8; ++r)
        sS[qi * 16 + r + 8 * half][col] = s[r] * 0.125f;
    }
    __syncthreads();

    // Online softmax, 4 threads per row.
    {
      float mx = -3.0e38f;
#pragma unroll
      for (int c = sseg * 16; c < sseg * 16 + 16; ++c)
        mx = fmaxf(mx, sS[srow][c]);
      sRed[srow][sseg] = mx;
    }
    __syncthreads();
    if (tid < 64) {
      float m2 = fmaxf(fmaxf(sRed[tid][0], sRed[tid][1]),
                       fmaxf(sRed[tid][2], sRed[tid][3]));
      m2 = fmaxf(m2, sM[tid]);
      sAl[tid] = __expf(sM[tid] - m2);
      sM[tid] = m2;
    }
    __syncthreads();
    {
      const float rm = sM[srow];
      float sum = 0.f;
#pragma unroll
      for (int c = sseg * 16; c < sseg * 16 + 16; ++c) {
        float e = __expf(sS[srow][c] - rm);
        sS[srow][c] = e;
        sum += e;
      }
      sRed[srow][sseg] = sum;
    }
    __syncthreads();
    if (tid < 64)
      sL[tid] = sL[tid] * sAl[tid] +
                (sRed[tid][0] + sRed[tid][1] + sRed[tid][2] + sRed[tid][3]);

    // Rescale running accumulators by alpha(row).
#pragma unroll
    for (int t = 0; t < 2; ++t)
#pragma unroll
      for (int r = 0; r < 8; ++r)
        accO[t][r] *= sAl[qi * 16 + r + 8 * half];

    // O += P @ V ; A fragment packed f32 -> bf16 on the fly.
#pragma unroll
    for (int tj = 0; tj < 2; ++tj) {
#pragma unroll
      for (int kk = 0; kk < 64; kk += 32) {
        FragU f;
        const int row = qi * 16 + lr;
#pragma unroll
        for (int j = 0; j < 8; ++j) {
          const int k = a_kmap(j, half) + kk;
          unsigned int lo = f2bf(sS[row][k]);
          unsigned int hi = f2bf(sS[row][k + 1]);
          f.w[j] = lo | (hi << 16);
        }
        v16bf bb = load_frag_b(&sV[cur][0][0], kk, 80, dj * 32 + tj * 16, lane);
        accO[tj] = wmma_bf16(f.v, bb, accO[tj]);
      }
    }
    // next iteration's top barrier orders buffer reuse and sS rewrite
  }
  __syncthreads();   // sL final values visible before normalization

  // Normalize and store heads in concat layout: A2[b][p][h*64+d]
#pragma unroll
  for (int tj = 0; tj < 2; ++tj) {
    const int d = dj * 32 + tj * 16 + lr;
#pragma unroll
    for (int r = 0; r < 8; ++r) {
      const int row = qi * 16 + r + 8 * half;
      const int p = p0 + row;
      const float v = accO[tj][r] / sL[row];
      A2[((size_t)(b * SEQ + p)) * DIMS + h * HEAD_DIM + d] = f2bf(v);
    }
  }
}

// ---------------------------------------------------------------------------
// Kernel 3: output projection (bf16 A2 x f32 Wp -> f32 out + bp).
// A tile staged via async DMA (bf16 source); W tile converted in-register.
// ---------------------------------------------------------------------------
__global__ void __launch_bounds__(256) out_proj_kernel(
    const unsigned short* __restrict__ A2,
    const float* __restrict__ Wp, const float* __restrict__ bp,
    float* __restrict__ out)
{
  __shared__ __align__(16) unsigned short sA[128][40];
  __shared__ __align__(16) unsigned short sB[32][130];

  const int tid = threadIdx.x, lane = tid & 31, wave = tid >> 5;
  const int wm = wave & 3, wn = wave >> 2;
  const int m0 = blockIdx.y * 128, n0 = blockIdx.x * 128;

  v8f acc[2][4] = {};

  for (int k0 = 0; k0 < DIMS; k0 += 32) {
    if (k0 + 32 < DIMS)
      __builtin_prefetch(Wp + (size_t)(k0 + 32 + (tid & 31)) * DIMS + n0, 0, 3);
#pragma unroll
    for (int i = 0; i < 2; ++i) {            // A tile 128x32, async 16B copies
      int e = tid + 256 * i;
      int r = e >> 2, c8 = (e & 3) * 8;
      async_copy_b128(&sA[r][c8], A2 + (size_t)(m0 + r) * DIMS + k0 + c8);
    }
#pragma unroll
    for (int i = 0; i < 16; ++i) {           // W tile 32x128, f32 -> bf16
      int e = tid + 256 * i;
      int kk = e >> 7, nn = e & 127;
      sB[kk][nn] = f2bf(Wp[(size_t)(k0 + kk) * DIMS + n0 + nn]);
    }
    wait_async0();
    __syncthreads();

    v16bf afr[2], bfr[4];
#pragma unroll
    for (int ti = 0; ti < 2; ++ti)
      afr[ti] = load_frag_a(&sA[0][0], wm * 32 + ti * 16 + (lane & 15), 40, 0, lane);
#pragma unroll
    for (int tj = 0; tj < 4; ++tj)
      bfr[tj] = load_frag_b(&sB[0][0], 0, 130, wn * 64 + tj * 16, lane);
#pragma unroll
    for (int ti = 0; ti < 2; ++ti)
#pragma unroll
      for (int tj = 0; tj < 4; ++tj)
        acc[ti][tj] = wmma_bf16(afr[ti], bfr[tj], acc[ti][tj]);
    __syncthreads();
  }

  const int half = lane >> 4, lr = lane & 15;
#pragma unroll
  for (int ti = 0; ti < 2; ++ti)
#pragma unroll
    for (int tj = 0; tj < 4; ++tj) {
      const int n = n0 + wn * 64 + tj * 16 + lr;
      const float bn = bp[n];
#pragma unroll
      for (int r = 0; r < 8; ++r) {
        const int m = m0 + wm * 32 + ti * 16 + r + 8 * half;
        out[(size_t)m * DIMS + n] = acc[ti][tj][r] + bn;
      }
    }
}

extern "C" void kernel_launch(void* const* d_in, const int* in_sizes, int n_in,
                              void* d_out, int out_size, void* d_ws, size_t ws_size,
                              hipStream_t stream) {
  (void)in_sizes; (void)n_in; (void)out_size; (void)ws_size;
  const float* q  = (const float*)d_in[0];
  const float* x  = (const float*)d_in[1];
  const float* Wq = (const float*)d_in[2];
  const float* bq = (const float*)d_in[3];
  const float* Wk = (const float*)d_in[4];
  const float* bk = (const float*)d_in[5];
  const float* Wv = (const float*)d_in[6];
  const float* bv = (const float*)d_in[7];
  const float* Wp = (const float*)d_in[8];
  const float* bp = (const float*)d_in[9];
  float* out = (float*)d_out;

  const size_t elems = (size_t)BATCH * SEQ * DIMS;      // 8M bf16 each
  unsigned short* Qs = (unsigned short*)d_ws;           // 16 MB
  unsigned short* Kt = Qs + elems;                      // 16 MB
  unsigned short* Vs = Kt + elems;                      // 16 MB
  unsigned short* A2 = Vs + elems;                      // 16 MB

  dim3 blk(256);
  qkv_proj_kernel<<<dim3(DIMS / 128, MTOT / 128, 3), blk, 0, stream>>>(
      q, x, Wq, bq, Wk, bk, Wv, bv, Qs, Kt, Vs);
  attention_kernel<<<dim3(SEQ / 64, NUM_HEADS, BATCH), blk, 0, stream>>>(
      Qs, Kt, Vs, A2);
  out_proj_kernel<<<dim3(DIMS / 128, MTOT / 128), blk, 0, stream>>>(
      A2, Wp, bp, out);
}
